// MultiFeatureAttention_21242908246364
// MI455X (gfx1250) — matrix operand order repaired and checked by reading, not compiled, not run on hardware
//
#include <hip/hip_runtime.h>

// MI455X / gfx1250, wave32. Problem is tiny (~0.6 GFLOP, ~4MB) -> latency bound.
// Masked softmax over (B,H,SQ,SK,DIM) collapses (mask depends only on b,k,d) into
// per-(b,h) GEMMs: out = (E @ mask*value) / (E @ mask). All dense GEMMs run on
// v_wmma_f32_16x16x32_bf16 (f32 accum); softmax/LN statistics stay in f32.
// v2: hw v_tanh/v_exp transcendentals, A-fragments hoisted out of N-tile loops,
//     2-lane-per-row softmax/LN with shfl_xor combines.

typedef __attribute__((ext_vector_type(16))) __bf16 v16bf;
typedef __attribute__((ext_vector_type(8)))  float  v8f;

__device__ __forceinline__ __bf16 f2bf(float f) { return (__bf16)f; }

__device__ __forceinline__ float fast_exp(float x) {   // exp(x), x <= 0 here
#if __has_builtin(__builtin_amdgcn_exp2f)
  return __builtin_amdgcn_exp2f(x * 1.4426950408889634f);
#else
  return exp2f(x * 1.4426950408889634f);
#endif
}

__device__ __forceinline__ float fast_tanh(float x) {
#if __has_builtin(__builtin_amdgcn_tanhf)
  return __builtin_amdgcn_tanhf(x);                    // CDNA5 V_TANH_F32
#else
  float e = fast_exp(-2.0f * fabsf(x));
  float t = (1.0f - e) / (1.0f + e);
  return x < 0.0f ? -t : t;
#endif
}

__device__ __forceinline__ v8f wmma_bf16(v16bf a, v16bf b, v8f c) {
  return __builtin_amdgcn_wmma_f32_16x16x32_bf16(false, a, false, b, (short)0, c,
                                                 false, false);
}

// A fragment (16x32 bf16, row-major LDS, row stride ldk elements).
// Contiguous k order (same as B frag): lane m=lane&15, hi=lane>>4: e_i = K(hi*16 + i) -> 16 contiguous bf16
__device__ __forceinline__ v16bf load_a_frag(const __bf16* base, int ldk, int lane) {
  const int m = lane & 15, hi = lane >> 4;
  const __bf16* p = base + m * ldk + hi * 16;
  v16bf a;
#pragma unroll
  for (int j = 0; j < 16; ++j) a[j] = p[j];
  return a;
}

// B fragment (32x16 KxN), stored TRANSPOSED [N][K] row-major in LDS (stride ldk).
// Contiguous k order (same as A frag): lane n=lane&15, hi=lane>>4: e_i = K(hi*16 + i) -> 16 contiguous bf16.
__device__ __forceinline__ v16bf load_b_fragT(const __bf16* baseT, int ldk, int lane) {
  const int n = lane & 15, hi = lane >> 4;
  const __bf16* p = baseT + n * ldk + hi * 16;
  v16bf b;
#pragma unroll
  for (int j = 0; j < 16; ++j) b[j] = p[j];
  return b;
}

// ---------------------------------------------------------------------------
// Kernel A: qe/ke = tanh(X@w1+b1)@w2 ; q/k = e@(wq|wk)+(bq|bk)
// grid = B*2 (b, which), 256 threads = 8 waves, each wave owns a 16-row tile.
// ---------------------------------------------------------------------------
__global__ __launch_bounds__(256) void k_embed(
    const float* __restrict__ qv, const float* __restrict__ kv,
    const float* __restrict__ w1, const float* __restrict__ b1,
    const float* __restrict__ w2,
    const float* __restrict__ wq, const float* __restrict__ bq,
    const float* __restrict__ wk, const float* __restrict__ bk,
    float* __restrict__ q_all, float* __restrict__ k_all)
{
  extern __shared__ char pool[];
  __bf16* XA  = (__bf16*)(pool);          // [128][32]   dead after stage 1
  __bf16* W1T = (__bf16*)(pool + 8192);   // [128][32]   dead after stage 1
  __bf16* T1  = (__bf16*)(pool + 16384);  // [128][128]  dead after stage 2
  __bf16* W2T = (__bf16*)(pool + 49152);  // [64][128]
  __bf16* E   = (__bf16*)(pool + 65536);  // [128][64]
  __bf16* WPT = (__bf16*)(pool);          // [64][64]    reuses XA region

  const int tid = threadIdx.x, lane = tid & 31, wave = tid >> 5;
  const int which = blockIdx.x & 1, b = blockIdx.x >> 1;
  const float* X = (which ? kv : qv) + b * (128 * 32);

  __builtin_prefetch(w2, 0, 0);
  __builtin_prefetch(which ? wk : wq, 0, 0);

  for (int i = tid; i < 128 * 32; i += 256) XA[i] = f2bf(X[i]);
  for (int i = tid; i < 128 * 32; i += 256) {
    int n = i >> 5, k = i & 31;
    W1T[i] = f2bf(w1[k * 128 + n]);                 // w1 is (32,128)
  }
  for (int i = tid; i < 64 * 128; i += 256) {
    int n = i >> 7, k = i & 127;
    W2T[i] = f2bf(w2[k * 64 + n]);                  // w2 is (128,64)
  }
  __syncthreads();

  const int row0 = wave * 16;
  const int n = lane & 15, hi = lane >> 4;

  // stage 1: T1 = tanh(X @ w1 + b1)   (128x32)@(32x128), single K step
  {
    const v16bf afr = load_a_frag(XA + row0 * 32, 32, lane);
#pragma unroll
    for (int nt = 0; nt < 8; ++nt) {
      v8f acc = {};
      acc = wmma_bf16(afr, load_b_fragT(W1T + (nt * 16) * 32, 32, lane), acc);
      int col = nt * 16 + n;
      float bias = b1[col];
#pragma unroll
      for (int j = 0; j < 8; ++j)
        T1[(row0 + hi * 8 + j) * 128 + col] = f2bf(fast_tanh(acc[j] + bias));
    }
  }
  __syncthreads();

  // stage projection weights into WPT (XA region is dead now)
  {
    const float* wp = which ? wk : wq;
    for (int i = tid; i < 64 * 64; i += 256) {
      int nn = i >> 6, k = i & 63;
      WPT[i] = f2bf(wp[k * 64 + nn]);               // wq/wk are (64,64)
    }
  }

  // stage 2: E = T1 @ w2   (128x128)@(128x64)
  {
    v16bf afr[4];
#pragma unroll
    for (int ks = 0; ks < 4; ++ks)
      afr[ks] = load_a_frag(T1 + row0 * 128 + ks * 32, 128, lane);
#pragma unroll
    for (int nt = 0; nt < 4; ++nt) {
      v8f acc = {};
#pragma unroll
      for (int ks = 0; ks < 4; ++ks)
        acc = wmma_bf16(afr[ks], load_b_fragT(W2T + (nt * 16) * 128 + ks * 32, 128, lane), acc);
      int col = nt * 16 + n;
#pragma unroll
      for (int j = 0; j < 8; ++j)
        E[(row0 + hi * 8 + j) * 64 + col] = f2bf(acc[j]);
    }
  }
  __syncthreads();

  // stage 3: proj = E @ (wq|wk) + bias   (128x64)@(64x64)
  const float* bp = which ? bk : bq;
  float* dst = which ? k_all : q_all;
  {
    v16bf afr[2];
#pragma unroll
    for (int ks = 0; ks < 2; ++ks)
      afr[ks] = load_a_frag(E + row0 * 64 + ks * 32, 64, lane);
#pragma unroll
    for (int nt = 0; nt < 4; ++nt) {
      v8f acc = {};
#pragma unroll
      for (int ks = 0; ks < 2; ++ks)
        acc = wmma_bf16(afr[ks], load_b_fragT(WPT + (nt * 16) * 64 + ks * 32, 64, lane), acc);
      int col = nt * 16 + n;
      float bias = bp[col];
#pragma unroll
      for (int j = 0; j < 8; ++j)
        dst[(b * 128 + row0 + hi * 8 + j) * 64 + col] = acc[j] + bias;
    }
  }
}

// ---------------------------------------------------------------------------
// Kernel B: per (b,h): scores -> row max -> exp(f32) -> bf16 E;
//           Out = E @ [mask*value | mask]; x = numer/denom. grid = B*H.
// ---------------------------------------------------------------------------
__global__ __launch_bounds__(256) void k_attn(
    const float* __restrict__ q_all, const float* __restrict__ k_all,
    const float* __restrict__ value, const int* __restrict__ emb_mask,
    float* __restrict__ xbuf)
{
  extern __shared__ char pool[];
  float*  S  = (float*)(pool);                         // [128][128] f32
  __bf16* Eb = (__bf16*)(pool + 65536);                // [128][128]
  __bf16* Wt = (__bf16*)(pool + 65536 + 32768);        // [64][128] transposed [N][K]
  float*  Qh = (float*)(pool + 65536 + 32768 + 16384); // [128][8]
  float*  Kh = Qh + 1024;                              // [128][8]

  const int tid = threadIdx.x, lane = tid & 31, wave = tid >> 5;
  const int b = blockIdx.x >> 3, h = blockIdx.x & 7;

  for (int i = tid; i < 1024; i += 256) {
    int r = i >> 3, d = i & 7;
    Qh[i] = q_all[(b * 128 + r) * 64 + h * 8 + d];
    Kh[i] = k_all[(b * 128 + r) * 64 + h * 8 + d];
  }
  for (int i = tid; i < 64 * 128; i += 256) {
    int nn = i >> 7, k = i & 127;
    int c = nn & 31;
    int mk = emb_mask[(b * 128 + k) * 32 + c];
    float wv = (nn < 32) ? (mk ? value[(b * 128 + k) * 32 + c] : 0.0f)
                         : (mk ? 1.0f : 0.0f);
    Wt[i] = f2bf(wv);
  }
  __syncthreads();

  // scores (K-dim is only 8 -> f32 VALU keeps full precision pre-softmax)
  const float scale = 0.35355339059327373f; // 1/sqrt(8)
  for (int i = tid; i < 128 * 128; i += 256) {
    int r = i >> 7, c = i & 127;
    float acc = 0.f;
#pragma unroll
    for (int d = 0; d < 8; ++d) acc += Qh[r * 8 + d] * Kh[c * 8 + d];
    S[i] = acc * scale;
  }
  __syncthreads();

  // per-row max + exp, 2 adjacent lanes per row, shfl_xor combine
  {
    const int r = tid >> 1, hf = tid & 1;
    const float* sr = S + r * 128 + hf * 64;
    float m = -3.0e38f;
    for (int c = 0; c < 64; ++c) m = fmaxf(m, sr[c]);
    m = fmaxf(m, __shfl_xor(m, 1, 32));
    __bf16* er = Eb + r * 128 + hf * 64;
    for (int c = 0; c < 64; ++c) er[c] = f2bf(fast_exp(sr[c] - m));
  }
  __syncthreads();

  // GEMM: [numer | denom](128x64) = E(128x128) @ Wt^T ; then divide in-lane
  const int row0 = wave * 16, n = lane & 15, hi = lane >> 4;
  v16bf afr[4];
#pragma unroll
  for (int ks = 0; ks < 4; ++ks)
    afr[ks] = load_a_frag(Eb + row0 * 128 + ks * 32, 128, lane);
  v8f accs[4];
#pragma unroll
  for (int nt = 0; nt < 4; ++nt) {
    v8f acc = {};
#pragma unroll
    for (int ks = 0; ks < 4; ++ks)
      acc = wmma_bf16(afr[ks], load_b_fragT(Wt + (nt * 16) * 128 + ks * 32, 128, lane), acc);
    accs[nt] = acc;
  }
#pragma unroll
  for (int nt = 0; nt < 2; ++nt) {
#pragma unroll
    for (int j = 0; j < 8; ++j) {
      float o = accs[nt][j] / fmaxf(accs[nt + 2][j], 1.0e-30f);
      int m = row0 + hi * 8 + j;
      int d = nt * 16 + n;
      xbuf[(b * 128 + m) * 256 + h * 32 + d] = o;   // x[b][q][h*32+d]
    }
  }
}

// ---------------------------------------------------------------------------
// Kernel C: y = x @ wo + bo   (1024x256)@(256x128). grid = 8 (128 rows each).
// ---------------------------------------------------------------------------
__global__ __launch_bounds__(256) void k_yproj(
    const float* __restrict__ xbuf, const float* __restrict__ wo,
    const float* __restrict__ bo, float* __restrict__ y)
{
  extern __shared__ char pool[];
  __bf16* XA  = (__bf16*)(pool);          // [128][256]
  __bf16* WoT = (__bf16*)(pool + 65536);  // [128][256] transposed [N][K]

  const int tid = threadIdx.x, lane = tid & 31, wave = tid >> 5;
  const int rbase = blockIdx.x * 128;

  __builtin_prefetch(wo, 0, 0);

  for (int i = tid; i < 128 * 256; i += 256) XA[i] = f2bf(xbuf[rbase * 256 + i]);
  for (int i = tid; i < 128 * 256; i += 256) {
    int n = i >> 8, k = i & 255;
    WoT[i] = f2bf(wo[k * 128 + n]);                 // wo is (256,128)
  }
  __syncthreads();

  const int row0 = wave * 16, n = lane & 15, hi = lane >> 4;
  v16bf afr[8];
#pragma unroll
  for (int ks = 0; ks < 8; ++ks)
    afr[ks] = load_a_frag(XA + row0 * 256 + ks * 32, 256, lane);
#pragma unroll
  for (int nt = 0; nt < 8; ++nt) {
    v8f acc = {};
#pragma unroll
    for (int ks = 0; ks < 8; ++ks)
      acc = wmma_bf16(afr[ks], load_b_fragT(WoT + (nt * 16) * 256 + ks * 32, 256, lane), acc);
    int col = nt * 16 + n;
    float bias = bo[col];
#pragma unroll
    for (int j = 0; j < 8; ++j)
      y[(rbase + row0 + hi * 8 + j) * 128 + col] = acc[j] + bias;
  }
}

// ---------------------------------------------------------------------------
// Kernel D: impute branch: h1 = imp@wd1+bd1 ; LN ; relu ; @wd2+bd2. grid = 8.
// ---------------------------------------------------------------------------
__global__ __launch_bounds__(256) void k_impute(
    const float* __restrict__ imp, const float* __restrict__ wd1,
    const float* __restrict__ bd1, const float* __restrict__ lng,
    const float* __restrict__ lnb, const float* __restrict__ wd2,
    const float* __restrict__ bd2, float* __restrict__ qd)
{
  extern __shared__ char pool[];
  float*  H1  = (float*)(pool);                    // [128][128] f32
  __bf16* IA  = (__bf16*)(pool + 65536);           // [128][32] (K padded 16->32)
  __bf16* W1T = (__bf16*)(pool + 65536 + 8192);    // [128][32] (K padded)
  __bf16* A2  = (__bf16*)(pool + 65536 + 16384);   // [128][128]
  __bf16* W2T = (__bf16*)(pool + 65536 + 49152);   // [128][128]

  const int tid = threadIdx.x, lane = tid & 31, wave = tid >> 5;
  const int rbase = blockIdx.x * 128;

  __builtin_prefetch(wd2, 0, 0);

  for (int i = tid; i < 128 * 32; i += 256) {
    int r = i >> 5, k = i & 31;
    IA[i] = f2bf(k < 16 ? imp[(rbase + r) * 16 + k] : 0.0f);
  }
  for (int i = tid; i < 128 * 32; i += 256) {
    int n = i >> 5, k = i & 31;
    W1T[i] = f2bf(k < 16 ? wd1[k * 128 + n] : 0.0f);  // wd1 is (16,128)
  }
  for (int i = tid; i < 128 * 128; i += 256) {
    int n = i >> 7, k = i & 127;
    W2T[i] = f2bf(wd2[k * 128 + n]);                  // wd2 is (128,128)
  }
  __syncthreads();

  const int row0 = wave * 16, n = lane & 15, hi = lane >> 4;

  // stage 1: H1 = imp @ wd1 + bd1  (K=32 incl. zero pad, single step)
  {
    const v16bf afr = load_a_frag(IA + row0 * 32, 32, lane);
#pragma unroll
    for (int nt = 0; nt < 8; ++nt) {
      v8f acc = {};
      acc = wmma_bf16(afr, load_b_fragT(W1T + (nt * 16) * 32, 32, lane), acc);
      int col = nt * 16 + n;
      float bias = bd1[col];
#pragma unroll
      for (int j = 0; j < 8; ++j)
        H1[(row0 + hi * 8 + j) * 128 + col] = acc[j] + bias;
    }
  }
  __syncthreads();

  // LayerNorm + ReLU in f32, 2 adjacent lanes per row, shfl_xor combines
  {
    const int r = tid >> 1, hf = tid & 1;
    float* hr = H1 + r * 128 + hf * 64;
    float s = 0.f;
    for (int c = 0; c < 64; ++c) s += hr[c];
    s += __shfl_xor(s, 1, 32);
    float mu = s * (1.0f / 128.0f);
    float v = 0.f;
    for (int c = 0; c < 64; ++c) { float d = hr[c] - mu; v += d * d; }
    v += __shfl_xor(v, 1, 32);
    float inv = rsqrtf(v * (1.0f / 128.0f) + 1e-5f);
    const float* lg = lng + hf * 64;
    const float* lb = lnb + hf * 64;
    __bf16* ar = A2 + r * 128 + hf * 64;
    for (int c = 0; c < 64; ++c) {
      float t = (hr[c] - mu) * inv * lg[c] + lb[c];
      ar[c] = f2bf(fmaxf(t, 0.0f));
    }
  }
  __syncthreads();

  // stage 2: qd = relu(ln) @ wd2 + bd2  (128x128)@(128x128)
  {
    v16bf afr[4];
#pragma unroll
    for (int ks = 0; ks < 4; ++ks)
      afr[ks] = load_a_frag(A2 + row0 * 128 + ks * 32, 128, lane);
#pragma unroll
    for (int nt = 0; nt < 8; ++nt) {
      v8f acc = {};
#pragma unroll
      for (int ks = 0; ks < 4; ++ks)
        acc = wmma_bf16(afr[ks], load_b_fragT(W2T + (nt * 16) * 128 + ks * 32, 128, lane), acc);
      int col = nt * 16 + n;
      float bias = bd2[col];
#pragma unroll
      for (int j = 0; j < 8; ++j)
        qd[(rbase + row0 + hi * 8 + j) * 128 + col] = acc[j] + bias;
    }
  }
}

// ---------------------------------------------------------------------------
extern "C" void kernel_launch(void* const* d_in, const int* in_sizes, int n_in,
                              void* d_out, int out_size, void* d_ws, size_t ws_size,
                              hipStream_t stream) {
  (void)in_sizes; (void)n_in; (void)out_size; (void)ws_size;
  // setup_inputs() dict order
  const float* qv  = (const float*)d_in[0];
  const float* kv  = (const float*)d_in[1];
  const float* val = (const float*)d_in[2];
  const float* imp = (const float*)d_in[3];
  const int*   msk = (const int*)  d_in[4];
  const float* w1  = (const float*)d_in[5];
  const float* b1  = (const float*)d_in[6];
  const float* w2  = (const float*)d_in[7];
  const float* wq  = (const float*)d_in[8];
  const float* bq  = (const float*)d_in[9];
  const float* wk  = (const float*)d_in[10];
  const float* bk  = (const float*)d_in[11];
  const float* wo  = (const float*)d_in[12];
  const float* bo  = (const float*)d_in[13];
  const float* wd1 = (const float*)d_in[14];
  const float* bd1 = (const float*)d_in[15];
  const float* lng = (const float*)d_in[16];
  const float* lnb = (const float*)d_in[17];
  const float* wd2 = (const float*)d_in[18];
  const float* bd2 = (const float*)d_in[19];

  float* out = (float*)d_out;
  float* y   = out;            // (8,128,128)
  float* qd  = out + 131072;   // (8,128,128)

  float* ws    = (float*)d_ws; // 1.5 MB used
  float* q_all = ws;           // 8*128*64
  float* k_all = ws + 65536;   // 8*128*64
  float* xbuf  = ws + 131072;  // 8*128*256

  k_embed <<<16, 256, 81920,  stream>>>(qv, kv, w1, b1, w2, wq, bq, wk, bk, q_all, k_all);
  k_attn  <<<64, 256, 122880, stream>>>(q_all, k_all, val, msk, xbuf);
  k_yproj <<<8,  256, 131072, stream>>>(xbuf, wo, bo, y);
  k_impute<<<8,  256, 147456, stream>>>(imp, wd1, bd1, lng, lnb, wd2, bd2, qd);
}
